// RNNModel_57483842289934
// MI455X (gfx1250) — compile-verified, matrix operand order checked by reading, not run on hardware
//
#include <hip/hip_runtime.h>
#include <hip/hip_bf16.h>

typedef _Float16 half_t;
typedef __attribute__((ext_vector_type(16))) _Float16 v16h;
typedef __attribute__((ext_vector_type(8)))  _Float16 v8h;
typedef __attribute__((ext_vector_type(8)))  float    v8f;

#define BB 64
#define TT 2048
#define II 128
#define HH 256
#define OO 10

// x_proj is stored in WMMA C-fragment layout:
//   frag[t][mt][wtile][lane][r]  (halves), mt = b/16, wtile = n/16
// so phase-1 stores and phase-2 loads are single b128 ops per lane.
#define FRAG_STRIDE_T ((size_t)4 * 16 * 32 * 8)   // halves per t

// ---------------------------------------------------------------------------
// Phase 1: x_proj = inputs . W_ih^T + (b_ih+b_hh), written as f16 fragments.
// One block per t; 8 waves; wave w owns N-tiles {w, w+8}; K=128 -> 4 WMMAs.
// ---------------------------------------------------------------------------
__global__ __launch_bounds__(256) void rnn_xproj_wmma(
    const float* __restrict__ inputs,   // [B][T][I]
    const float* __restrict__ W_ih,     // [H][I]
    const float* __restrict__ b_ih,
    const float* __restrict__ b_hh,
    half_t* __restrict__ xproj)         // fragment layout
{
    const int t    = blockIdx.x;
    const int tid  = threadIdx.x;
    const int wave = tid >> 5;          // 0..7
    const int lane = tid & 31;
    const int lm   = lane & 15;
    const int hi   = lane >> 4;
    const int kb   = hi * 8;

    // B fragments: 2 N-tiles x 4 K-chunks, built from rows of W_ih (contiguous)
    v16h bfrag[2][4];
    float bias[2];
#pragma unroll
    for (int nt = 0; nt < 2; ++nt) {
        const int n = nt * 128 + wave * 16 + lm;
        const float* wrow = W_ih + (size_t)n * II;
#pragma unroll
        for (int kk = 0; kk < 4; ++kk) {
            v16h f;
#pragma unroll
            for (int h = 0; h < 8; ++h) f[h]     = (half_t)wrow[kk * 32 + kb + h];
#pragma unroll
            for (int h = 0; h < 8; ++h) f[8 + h] = (half_t)wrow[kk * 32 + 16 + kb + h];
            bfrag[nt][kk] = f;
        }
        bias[nt] = b_ih[n] + b_hh[n];
    }

#pragma unroll 1
    for (int mt = 0; mt < 4; ++mt) {
        // A fragments: lane lm = row m, two 8-half runs at kb and kb+16
        const int m = mt * 16 + lm;
        const float* arow = inputs + ((size_t)m * TT + t) * II;
        v16h afrag[4];
#pragma unroll
        for (int kk = 0; kk < 4; ++kk) {
            v16h f;
#pragma unroll
            for (int h = 0; h < 8; ++h) f[h]     = (half_t)arow[kk * 32 + kb + h];
#pragma unroll
            for (int h = 0; h < 8; ++h) f[8 + h] = (half_t)arow[kk * 32 + 16 + kb + h];
            afrag[kk] = f;
        }
#pragma unroll
        for (int nt = 0; nt < 2; ++nt) {
            v8f acc = {};
#pragma unroll
            for (int kk = 0; kk < 4; ++kk)
                acc = __builtin_amdgcn_wmma_f32_16x16x32_f16(
                    false, afrag[kk], false, bfrag[nt][kk], (short)0, acc, false, false);
            // Store as one b128 fragment per lane (coalesced 512B per tile)
            const int wtile = nt * 8 + wave;
            v8h pk;
#pragma unroll
            for (int r = 0; r < 8; ++r) pk[r] = (half_t)(acc[r] + bias[nt]);
            *(v8h*)(xproj + ((((size_t)t * 4 + mt) * 16 + wtile) * 32 + lane) * 8) = pk;
        }
    }
}

// ---------------------------------------------------------------------------
// Phase 2: sequential scan. 4 blocks (16 batch rows each), 16 waves.
// - W_hh^T B-fragments register-resident (loop-invariant)
// - f32 h lives in registers (fragment ownership is stable across steps)
// - f16 h double-buffered in LDS (broadcast A-fragments for all waves)
// - x_proj fragment loads software-pipelined one step ahead
// - two WMMA accumulators to break the RAW hazard chain
// ---------------------------------------------------------------------------
__global__ __launch_bounds__(512) void rnn_scan_wmma(
    const half_t* __restrict__ xproj,   // fragment layout
    const float* __restrict__ W_hh,     // [H][H]
    float* __restrict__ h_final)        // [B][H]
{
    __shared__ half_t h16[2][16][HH];

    const int bk   = blockIdx.x;        // batch tile 0..3
    const int b0   = bk * 16;
    const int tid  = threadIdx.x;
    const int wave = tid >> 5;          // 0..15 -> N-tile
    const int lane = tid & 31;
    const int lm   = lane & 15;
    const int hi   = lane >> 4;
    const int kb   = hi * 8;
    const int n    = wave * 16 + lm;

    // Register-resident B fragments: B[k][n] = W_hh[n][k] (contiguous reads)
    v16h bfrag[8];
    {
        const float* wrow = W_hh + (size_t)n * HH;
#pragma unroll
        for (int kk = 0; kk < 8; ++kk) {
            v16h f;
#pragma unroll
            for (int h = 0; h < 8; ++h) f[h]     = (half_t)wrow[kk * 32 + kb + h];
#pragma unroll
            for (int h = 0; h < 8; ++h) f[8 + h] = (half_t)wrow[kk * 32 + 16 + kb + h];
            bfrag[kk] = f;
        }
    }

    // h0 = 0 : registers + LDS f16 buffer 0
    float hreg[8];
#pragma unroll
    for (int r = 0; r < 8; ++r) hreg[r] = 0.0f;
    {
        half_t* p16 = &h16[0][0][0];
        for (int idx = tid; idx < 2 * 16 * HH; idx += blockDim.x)
            p16[idx] = (half_t)0.0f;
    }
    __syncthreads();

    // x_proj fragment pointer for this (block, wave, lane)
    const half_t* fp = xproj + (((size_t)bk * 16 + wave) * 32 + lane) * 8;
    v8h xq = *(const v8h*)fp;           // fragment for t = 0

    const float invT = 1.0f / (float)TT;
    int cur = 0;
#pragma unroll 1
    for (int t = 0; t < TT; ++t) {
        const int nxt = cur ^ 1;

        // Prefetch next step's x_proj fragment (one b128, consumed next iter)
        const int tn = (t + 1 < TT) ? (t + 1) : t;
        const v8h xnext = *(const v8h*)(fp + (size_t)tn * FRAG_STRIDE_T);

        // h @ W_hh^T : A-fragments from LDS, 8 WMMAs on two accumulators
        v8f acc0 = {}, acc1 = {};
#pragma unroll
        for (int kk = 0; kk < 8; ++kk) {
            const half_t* row = &h16[cur][lm][kk * 32 + kb];
            v16h f;
#pragma unroll
            for (int h = 0; h < 8; ++h) f[h]     = row[h];
#pragma unroll
            for (int h = 0; h < 8; ++h) f[8 + h] = row[16 + h];
            if (kk & 1)
                acc1 = __builtin_amdgcn_wmma_f32_16x16x32_f16(
                    false, f, false, bfrag[kk], (short)0, acc1, false, false);
            else
                acc0 = __builtin_amdgcn_wmma_f32_16x16x32_f16(
                    false, f, false, bfrag[kk], (short)0, acc0, false, false);
        }

        // h_new = tanh(xp + mm) * invT + h_old  (fast tanh: exp2 + rcp, clamped)
#pragma unroll
        for (int r = 0; r < 8; ++r) {
            const int m = r + 8 * hi;
            float pre = (float)xq[r] + acc0[r] + acc1[r];
            pre = fminf(fmaxf(pre, -20.0f), 20.0f);
            const float e  = __builtin_amdgcn_exp2f(pre * 2.8853900817779268f); // e^(2x)
            const float th = (e - 1.0f) * __builtin_amdgcn_rcpf(e + 1.0f);
            const float hn = th * invT + hreg[r];
            hreg[r] = hn;
            h16[nxt][m][n] = (half_t)hn;
        }
        __syncthreads();
        cur = nxt;
        xq  = xnext;
    }

    // Export final h (each lane owns its 8 elements)
#pragma unroll
    for (int r = 0; r < 8; ++r) {
        const int m = r + 8 * hi;
        h_final[(size_t)(b0 + m) * HH + n] = hreg[r];
    }
}

// ---------------------------------------------------------------------------
// Phase 3: out = h_final @ W_out^T + b_out   (64 x 10, trivial)
// ---------------------------------------------------------------------------
__global__ __launch_bounds__(640) void rnn_head(
    const float* __restrict__ h_final,  // [B][H]
    const float* __restrict__ W_out,    // [O][H]
    const float* __restrict__ b_out,    // [O]
    float* __restrict__ out)            // [B][O]
{
    const int tid = threadIdx.x;
    if (tid >= BB * OO) return;
    const int b = tid / OO;
    const int o = tid % OO;
    const float* hv = h_final + (size_t)b * HH;
    const float* wv = W_out + (size_t)o * HH;
    float s = b_out[o];
#pragma unroll 8
    for (int k = 0; k < HH; ++k) s += hv[k] * wv[k];
    out[b * OO + o] = s;
}

extern "C" void kernel_launch(void* const* d_in, const int* in_sizes, int n_in,
                              void* d_out, int out_size, void* d_ws, size_t ws_size,
                              hipStream_t stream) {
    (void)in_sizes; (void)n_in; (void)out_size; (void)ws_size;
    const float* inputs = (const float*)d_in[0];   // [64][2048][128]
    const float* W_ih   = (const float*)d_in[1];   // [256][128]
    const float* W_hh   = (const float*)d_in[2];   // [256][256]
    const float* b_ih   = (const float*)d_in[3];   // [256]
    const float* b_hh   = (const float*)d_in[4];   // [256]
    const float* W_out  = (const float*)d_in[5];   // [10][256]
    const float* b_out  = (const float*)d_in[6];   // [10]
    float* out = (float*)d_out;                    // [64][10]

    half_t* xproj  = (half_t*)d_ws;                                       // 64 MB
    float*  hfinal = (float*)((char*)d_ws + (size_t)TT * FRAG_STRIDE_T * sizeof(half_t));

    rnn_xproj_wmma<<<TT, 256, 0, stream>>>(inputs, W_ih, b_ih, b_hh, xproj);
    rnn_scan_wmma<<<BB / 16, 512, 0, stream>>>(xproj, W_hh, hfinal);
    rnn_head<<<1, 640, 0, stream>>>(hfinal, W_out, b_out, out);
}
